// BatchRNN_84602265796641
// MI455X (gfx1250) — matrix operand order repaired and checked by reading, not compiled, not run on hardware
//
#include <hip/hip_runtime.h>
#include <cstdint>
#include <cstddef>

// Problem constants (match reference)
#define BB   32
#define TT   512
#define CC   768
#define HH   768
#define G4   3072   // 4*H
#define NG   32     // groups
#define CPG  24     // channels per group
#define EPSV 1e-5f

// ---------------- CDNA5 WMMA types ----------------
typedef __attribute__((ext_vector_type(16))) __bf16    v16bf;
typedef __attribute__((ext_vector_type(8)))  float     v8f;
typedef __attribute__((ext_vector_type(4)))  uint32_t  u32x4;
typedef __attribute__((ext_vector_type(8)))  uint32_t  u32x8;
typedef __attribute__((ext_vector_type(4)))  int       v4i;

union FragU { u32x8 u; v16bf v; };

// CDNA5 async global->LDS path (guarded; falls back to vec copy)
#if defined(__has_builtin)
#if __has_builtin(__builtin_amdgcn_global_load_async_to_lds_b128)
#define HAVE_ASYNC_LDS 1
#endif
#endif

#if defined(HAVE_ASYNC_LDS)
typedef __attribute__((address_space(1))) v4i* as1_v4i;
typedef __attribute__((address_space(3))) v4i* as3_v4i;
#endif

__device__ __forceinline__ void wait_asynccnt0() {
#if defined(__has_builtin) && __has_builtin(__builtin_amdgcn_s_wait_asynccnt)
  __builtin_amdgcn_s_wait_asynccnt(0);
#else
  asm volatile("s_wait_asynccnt 0" ::: "memory");
#endif
}

__device__ __forceinline__ unsigned short f2bf(float f) {
  uint32_t u = __float_as_uint(f);
  uint32_t r = u + 0x7FFFu + ((u >> 16) & 1u);   // round-to-nearest-even
  return (unsigned short)(r >> 16);
}

// A-operand fragment: 16-bit A matrix 16x32 (MxK), ISA 7.12.2 layout.
// lane L (L<16): row m0+L, K = {0..7, 16..23}; lane L+16: same rows, K = {8..15, 24..31}
__device__ __forceinline__ v16bf load_a_frag(const unsigned short* A,
                                             int ldk, int m0, int k0, int lane) {
  const int half = lane >> 4;
  const unsigned short* p = A + (size_t)(m0 + (lane & 15)) * ldk + k0 + half * 8;
  u32x4 q0 = *(const u32x4*)p;          // K = base .. base+7
  u32x4 q1 = *(const u32x4*)(p + 16);   // K = base+16 .. base+23
  FragU f;
  f.u[0] = q0[0]; f.u[1] = q0[1]; f.u[2] = q0[2]; f.u[3] = q0[3];
  f.u[4] = q1[0]; f.u[5] = q1[1]; f.u[6] = q1[2]; f.u[7] = q1[3];
  return f.v;
}

// B-operand fragment: 16-bit B matrix 32x16 (KxN) where B = W^T, W is [N,K] row-major.
// lane L: column n0+(L&15); lanes 0-15 hold K = {0..7, 16..23}, lanes 16-31 hold K = {8..15, 24..31}
__device__ __forceinline__ v16bf load_b_frag(const unsigned short* __restrict__ W,
                                             int ldk, int n0, int k0, int lane) {
  const int half = lane >> 4;
  const unsigned short* p = W + (size_t)(n0 + (lane & 15)) * ldk + k0 + half * 8;
  u32x4 q0 = *(const u32x4*)p;          // K = base .. base+7
  u32x4 q1 = *(const u32x4*)(p + 16);   // K = base+16 .. base+23
  FragU f;
  f.u[0] = q0[0]; f.u[1] = q0[1]; f.u[2] = q0[2]; f.u[3] = q0[3];
  f.u[4] = q1[0]; f.u[5] = q1[1]; f.u[6] = q1[2]; f.u[7] = q1[3];
  return f.v;
}

__device__ __forceinline__ v8f wmma_bf16(v16bf a, v16bf b, v8f c) {
  return __builtin_amdgcn_wmma_f32_16x16x32_bf16(false, a, false, b, (short)0, c,
                                                 false, false);
}

// ---------------- GroupNorm ----------------
__global__ __launch_bounds__(256)
void gn_stats_kernel(const float* __restrict__ x, float* __restrict__ meanp,
                     float* __restrict__ rstdp) {
  __shared__ float ssum[256];
  __shared__ float ssq[256];
  const int tid = threadIdx.x;
  const int b = blockIdx.x >> 5;
  const int g = blockIdx.x & 31;
  const float* xb = x + (size_t)b * TT * CC + g * CPG;
  float s = 0.f, sq = 0.f;
  for (int e = tid; e < TT * CPG; e += 256) {
    int t = e / CPG, ch = e - t * CPG;
    float v = xb[(size_t)t * CC + ch];
    s += v; sq += v * v;
  }
  ssum[tid] = s; ssq[tid] = sq;
  __syncthreads();
  for (int o = 128; o > 0; o >>= 1) {
    if (tid < o) { ssum[tid] += ssum[tid + o]; ssq[tid] += ssq[tid + o]; }
    __syncthreads();
  }
  if (tid == 0) {
    const float inv_n = 1.f / (float)(TT * CPG);
    float m = ssum[0] * inv_n;
    float var = ssq[0] * inv_n - m * m;
    meanp[blockIdx.x] = m;
    rstdp[blockIdx.x] = rsqrtf(var + EPSV);
  }
}

// Apply GN, write bf16 xn in [T,B,C] layout (K-contiguous rows for GEMM A)
__global__ __launch_bounds__(256)
void gn_apply_kernel(const float* __restrict__ x, const float* __restrict__ gamma,
                     const float* __restrict__ beta, const float* __restrict__ meanp,
                     const float* __restrict__ rstdp, unsigned short* __restrict__ xn) {
  size_t e = (size_t)blockIdx.x * 256 + threadIdx.x;  // over B*T*C
  int c = (int)(e % CC);
  int t = (int)((e / CC) % TT);
  int b = (int)(e / ((size_t)CC * TT));
  int g = c / CPG;
  float m = meanp[b * NG + g];
  float r = rstdp[b * NG + g];
  float v = (x[e] - m) * r * gamma[c] + beta[c];
  xn[((size_t)t * BB + b) * CC + c] = f2bf(v);
}

// ---------------- small utility kernels ----------------
__global__ __launch_bounds__(256)
void conv_bf16_kernel(const float* __restrict__ in, unsigned short* __restrict__ out, int n) {
  int e = blockIdx.x * 256 + threadIdx.x;
  if (e < n) out[e] = f2bf(in[e]);
}

__global__ __launch_bounds__(256)
void bias_sum_kernel(const float* __restrict__ a, const float* __restrict__ b,
                     float* __restrict__ out, int n) {
  int e = blockIdx.x * 256 + threadIdx.x;
  if (e < n) out[e] = a[e] + b[e];
}

__global__ __launch_bounds__(256)
void init_state_kernel(unsigned short* __restrict__ h, float* __restrict__ c, int n) {
  int e = blockIdx.x * 256 + threadIdx.x;
  if (e < n) { h[e] = 0; c[e] = 0.f; }
}

// ---------------- big input GEMM: Out[M,N] = A[M,K] * W[N,K]^T + bias[N] ----------------
// grid = (N/256, M/32), block = 256 (8 waves). Wave: 16x64 strip (4 accumulators).
// Plain loop body: the backend unrolls and software-pipelines the loads itself
// (manual register double-buffering generated v_mov storms — reverted).
__global__ __launch_bounds__(256)
void gemm_xw_kernel(const unsigned short* __restrict__ A,
                    const unsigned short* __restrict__ W,
                    const float* __restrict__ bias,
                    float* __restrict__ Out,
                    int M, int N, int K) {
  const int lane = threadIdx.x & 31;
  const int wave = threadIdx.x >> 5;
  const int mt = wave & 1;
  const int nt = wave >> 1;                      // 0..3
  const int m0 = blockIdx.y * 32 + mt * 16;
  const int n0 = blockIdx.x * 256 + nt * 64;

  v8f acc[4];
#pragma unroll
  for (int j = 0; j < 4; ++j) acc[j] = (v8f){};

  for (int k0 = 0; k0 < K; k0 += 32) {
    v16bf a = load_a_frag(A, K, m0, k0, lane);
    if (k0 + 32 < K)  // hint next A chunk into cache: global_prefetch_b8
      __builtin_prefetch(A + (size_t)(m0 + (lane & 15)) * K + k0 + 32, 0, 1);
#pragma unroll
    for (int j = 0; j < 4; ++j) {
      v16bf b = load_b_frag(W, K, n0 + j * 16, k0, lane);
      acc[j] = wmma_bf16(a, b, acc[j]);
    }
  }

  const int half = lane >> 4;
  const int col = lane & 15;
#pragma unroll
  for (int j = 0; j < 4; ++j) {
    const int c = n0 + j * 16 + col;
    const float bi = bias[c];
#pragma unroll
    for (int r = 0; r < 8; ++r) {
      const int row = m0 + r + half * 8;
      Out[(size_t)row * N + c] = acc[j][r] + bi;
    }
  }
}

// ---------------- fused recurrent step (both directions) ----------------
// grid = (H/64, 2 dirs), block = 256 (8 waves).
// Phase 1: stage h[dir] (32x768 bf16, 48KB) into LDS via async-to-LDS.
// Phase 2: WMMA; wave w owns gate region w>>1 and 32-col half (w&1): 2 m-tiles x
//          2 n-tiles, so every W row is fetched exactly once per block.
// Phase 3: gates staged in (reused) LDS; Phase 4: LSTM cell fused.
__global__ __launch_bounds__(256)
void lstm_step_kernel(const unsigned short* __restrict__ h_in,   // [2][B][H] bf16
                      unsigned short* __restrict__ h_out,        // [2][B][H] bf16
                      const unsigned short* __restrict__ whh_f,  // [4H][H] bf16
                      const unsigned short* __restrict__ whh_b,  // [4H][H] bf16
                      const float* __restrict__ xg_f,            // [T][B][4H]
                      const float* __restrict__ xg_b,            // [T][B][4H]
                      float* __restrict__ c_state,               // [2][B][H]
                      float* __restrict__ Out,                   // [B][T][2H]
                      int s) {
  __shared__ __align__(16) unsigned short smem[BB * HH];  // 48KB; reused for gates
  float* lds_g = (float*)smem;                            // 4*32*64 fp32 = 32KB

  const int dir = blockIdx.y;
  const int t = dir ? (TT - 1 - s) : s;
  const int ns = blockIdx.x * 64;                // column slice within H
  const int lane = threadIdx.x & 31;
  const int wave = threadIdx.x >> 5;
  const int reg = wave >> 1;                     // gate region 0..3 (i,f,g,o)
  const int cb = (wave & 1) * 32;                // 32-col half within slice
  const unsigned short* Wm = dir ? whh_b : whh_f;
  const unsigned short* hbase = h_in + dir * (BB * HH);

  // ---- Phase 1: cooperative h -> LDS (3072 x 16B chunks, 12 per thread) ----
  {
    const u32x4* src = (const u32x4*)hbase;
    u32x4* dst = (u32x4*)smem;
#if defined(HAVE_ASYNC_LDS)
    for (int i = threadIdx.x; i < (BB * HH * 2) / 16; i += 256) {
      __builtin_amdgcn_global_load_async_to_lds_b128(
          (as1_v4i)(u32x4*)(src + i), (as3_v4i)(dst + i), 0, 0);
    }
    wait_asynccnt0();
#else
    for (int i = threadIdx.x; i < (BB * HH * 2) / 16; i += 256) dst[i] = src[i];
#endif
  }
  __syncthreads();

  // ---- Phase 2: gates[reg][0:32][cb:cb+32] = h @ W^T ----
  v8f acc00 = (v8f){}, acc01 = (v8f){}, acc10 = (v8f){}, acc11 = (v8f){};
  const int nbase = reg * HH + ns + cb;          // global W row base for this wave
  const unsigned short* hs = smem;               // LDS-resident A matrix

  for (int k0 = 0; k0 < HH; k0 += 32) {
    v16bf a0 = load_a_frag(hs, HH, 0, k0, lane);    // ds_load_b128
    v16bf a1 = load_a_frag(hs, HH, 16, k0, lane);
    v16bf b0 = load_b_frag(Wm, HH, nbase, k0, lane);
    v16bf b1 = load_b_frag(Wm, HH, nbase + 16, k0, lane);
    acc00 = wmma_bf16(a0, b0, acc00);
    acc01 = wmma_bf16(a0, b1, acc01);
    acc10 = wmma_bf16(a1, b0, acc10);
    acc11 = wmma_bf16(a1, b1, acc11);
  }
  __syncthreads();   // everyone done reading h from LDS before gate overwrite

  // ---- Phase 3: stage gate pre-activations into (reused) LDS ----
  const int half = lane >> 4;
  const int col = lane & 15;
#pragma unroll
  for (int r = 0; r < 8; ++r) {
    const int row0 = r + half * 8;
    const int row1 = 16 + r + half * 8;
    lds_g[(reg * 32 + row0) * 64 + cb + col]      = acc00[r];
    lds_g[(reg * 32 + row0) * 64 + cb + 16 + col] = acc01[r];
    lds_g[(reg * 32 + row1) * 64 + cb + col]      = acc10[r];
    lds_g[(reg * 32 + row1) * 64 + cb + 16 + col] = acc11[r];
  }
  __syncthreads();

  // ---- Phase 4: LSTM cell on the 32x64 slice ----
  const float* xg = (dir ? xg_b : xg_f) + (size_t)t * (BB * G4);
  float* cst = c_state + dir * (BB * HH);
  unsigned short* ho = h_out + dir * (BB * HH);

  for (int e = threadIdx.x; e < 32 * 64; e += 256) {
    const int b = e >> 6;
    const int j = e & 63;
    const int n = ns + j;
    const size_t xo = (size_t)b * G4 + n;
    float iv = lds_g[(0 * 32 + b) * 64 + j] + xg[xo];
    float fv = lds_g[(1 * 32 + b) * 64 + j] + xg[xo + HH];
    float gv = lds_g[(2 * 32 + b) * 64 + j] + xg[xo + 2 * HH];
    float ov = lds_g[(3 * 32 + b) * 64 + j] + xg[xo + 3 * HH];
    iv = 1.f / (1.f + __expf(-iv));
    fv = 1.f / (1.f + __expf(-fv));
    gv = tanhf(gv);
    ov = 1.f / (1.f + __expf(-ov));
    const int ci = b * HH + n;
    const float cn = fv * cst[ci] + iv * gv;
    cst[ci] = cn;
    const float h = ov * tanhf(cn);
    ho[ci] = f2bf(h);
    Out[((size_t)b * TT + t) * (2 * HH) + dir * HH + n] = h;
  }
}

// ---------------- host launcher ----------------
extern "C" void kernel_launch(void* const* d_in, const int* in_sizes, int n_in,
                              void* d_out, int out_size, void* d_ws, size_t ws_size,
                              hipStream_t stream) {
  (void)in_sizes; (void)n_in; (void)out_size; (void)ws_size;

  const float* x      = (const float*)d_in[0];
  const float* gamma  = (const float*)d_in[1];
  const float* beta   = (const float*)d_in[2];
  const float* w_ih_f = (const float*)d_in[3];
  const float* w_hh_f = (const float*)d_in[4];
  const float* b_ih_f = (const float*)d_in[5];
  const float* b_hh_f = (const float*)d_in[6];
  const float* w_ih_b = (const float*)d_in[7];
  const float* w_hh_b = (const float*)d_in[8];
  const float* b_ih_b = (const float*)d_in[9];
  const float* b_hh_b = (const float*)d_in[10];
  float* out = (float*)d_out;

  // workspace carve-up (256B aligned slices)
  char* ws = (char*)d_ws;
  size_t off = 0;
  auto take = [&](size_t bytes) -> void* {
    void* p = ws + off;
    off = (off + bytes + 255) & ~(size_t)255;
    return p;
  };
  float*          meanp   = (float*)take((size_t)BB * NG * 4);
  float*          rstdp   = (float*)take((size_t)BB * NG * 4);
  unsigned short* xn      = (unsigned short*)take((size_t)TT * BB * CC * 2);
  unsigned short* wihf_bf = (unsigned short*)take((size_t)G4 * CC * 2);
  unsigned short* whhf_bf = (unsigned short*)take((size_t)G4 * HH * 2);
  unsigned short* wihb_bf = (unsigned short*)take((size_t)G4 * CC * 2);
  unsigned short* whhb_bf = (unsigned short*)take((size_t)G4 * HH * 2);
  float*          biasf   = (float*)take((size_t)G4 * 4);
  float*          biasb   = (float*)take((size_t)G4 * 4);
  float*          xgf     = (float*)take((size_t)TT * BB * G4 * 4);
  float*          xgb     = (float*)take((size_t)TT * BB * G4 * 4);
  unsigned short* hA      = (unsigned short*)take((size_t)2 * BB * HH * 2);
  unsigned short* hB      = (unsigned short*)take((size_t)2 * BB * HH * 2);
  float*          cst     = (float*)take((size_t)2 * BB * HH * 4);

  // 1) GroupNorm
  gn_stats_kernel<<<BB * NG, 256, 0, stream>>>(x, meanp, rstdp);
  gn_apply_kernel<<<((size_t)BB * TT * CC) / 256, 256, 0, stream>>>(
      x, gamma, beta, meanp, rstdp, xn);

  // 2) weight/bias prep
  const int wn = G4 * CC;
  conv_bf16_kernel<<<(wn + 255) / 256, 256, 0, stream>>>(w_ih_f, wihf_bf, wn);
  conv_bf16_kernel<<<(wn + 255) / 256, 256, 0, stream>>>(w_hh_f, whhf_bf, wn);
  conv_bf16_kernel<<<(wn + 255) / 256, 256, 0, stream>>>(w_ih_b, wihb_bf, wn);
  conv_bf16_kernel<<<(wn + 255) / 256, 256, 0, stream>>>(w_hh_b, whhb_bf, wn);
  bias_sum_kernel<<<(G4 + 255) / 256, 256, 0, stream>>>(b_ih_f, b_hh_f, biasf, G4);
  bias_sum_kernel<<<(G4 + 255) / 256, 256, 0, stream>>>(b_ih_b, b_hh_b, biasb, G4);

  // 3) initial h (bf16) and c state = 0
  init_state_kernel<<<(2 * BB * HH + 255) / 256, 256, 0, stream>>>(hA, cst, 2 * BB * HH);

  // 4) big input GEMMs: xg = xn @ w_ih^T + (b_ih + b_hh)
  const int M = TT * BB;   // 16384
  gemm_xw_kernel<<<dim3(G4 / 256, M / 32), 256, 0, stream>>>(xn, wihf_bf, biasf, xgf, M, G4, CC);
  gemm_xw_kernel<<<dim3(G4 / 256, M / 32), 256, 0, stream>>>(xn, wihb_bf, biasb, xgb, M, G4, CC);

  // 5) recurrence: 512 fused steps, h double-buffered
  for (int s = 0; s < TT; ++s) {
    const unsigned short* hin = (s & 1) ? hB : hA;
    unsigned short* hout = (s & 1) ? hA : hB;
    lstm_step_kernel<<<dim3(HH / 64, 2), 256, 0, stream>>>(
        hin, hout, whhf_bf, whhb_bf, xgf, xgb, cst, out, s);
  }
}
